// BasicGRU_22187801051906
// MI455X (gfx1250) — matrix operand order repaired and checked by reading, not compiled, not run on hardware
//
#include <hip/hip_runtime.h>
#include <hip/hip_bf16.h>
#include <stdint.h>

#define B_  128
#define T_  1024
#define I_  256
#define H_  512
#define IH_ 768   // I + H

#define NBLOCKS 32
#define WPAD    776   // padded LDS row stride (elements): 1552 B, bank-conflict-free
#define SMEM_BYTES ((32 + 16) * WPAD * 2)   // 74,496 B (< 320 KB WGP LDS)

typedef __attribute__((ext_vector_type(16))) __bf16 v16bf;
typedef __attribute__((ext_vector_type(8)))  float  v8f;

union Frag16 {
  v16bf v;
  uint4 q[2];
};

static __device__ __forceinline__ unsigned short f2bf(float f) {
  unsigned u = __float_as_uint(f);
  u += 0x7FFFu + ((u >> 16) & 1u);      // round-to-nearest-even
  return (unsigned short)(u >> 16);
}

static __device__ __forceinline__ float sigmoid_fast(float x) {
  return 1.0f / (1.0f + __expf(-x));
}
static __device__ __forceinline__ float tanh_fast(float x) {
  float e = __expf(2.0f * x);
  return (e - 1.0f) / (e + 1.0f);
}

// Monotonic device-wide barrier: counter only increases; barrier k waits until
// counter >= nblocks*k. prep_state_kernel re-zeros it every launch.
static __device__ __forceinline__ void grid_sync(unsigned* bar, unsigned target) {
  __threadfence();
  __syncthreads();
  if (threadIdx.x == 0) {
    atomicAdd(bar, 1u);
    while (atomicAdd(bar, 0u) < target) __builtin_amdgcn_s_sleep(2);
  }
  __syncthreads();
  __threadfence();
}

// ---------------- prep kernels ----------------

__global__ void prep_x_kernel(const float* __restrict__ x,
                              unsigned short* __restrict__ xb, size_t n) {
  size_t i = (size_t)blockIdx.x * blockDim.x + threadIdx.x;
  size_t stride = (size_t)gridDim.x * blockDim.x;
  for (; i < n; i += stride) xb[i] = f2bf(x[i]);
}

// Wz,Wr,Wh are [IH_][H_] row-major (K-major). Emit transposed bf16 [N][K]:
// Wzr_t: [1024][768] with N<512 -> Wz, N>=512 -> Wr.  Wh_t: [512][768].
__global__ void prep_w_kernel(const float* __restrict__ Wz,
                              const float* __restrict__ Wr,
                              const float* __restrict__ Wh,
                              unsigned short* __restrict__ Wzr_t,
                              unsigned short* __restrict__ Wh_t) {
  int i = blockIdx.x * blockDim.x + threadIdx.x;
  const int nzr = 1024 * IH_;
  const int nh  = H_ * IH_;
  if (i < nzr) {
    int n = i / IH_, k = i % IH_;
    float v = (n < H_) ? Wz[(size_t)k * H_ + n] : Wr[(size_t)k * H_ + (n - H_)];
    Wzr_t[i] = f2bf(v);
  } else if (i < nzr + nh) {
    int j = i - nzr;
    int n = j / IH_, k = j % IH_;
    Wh_t[j] = f2bf(Wh[(size_t)k * H_ + n]);
  }
}

__global__ void prep_state_kernel(unsigned short* h_bf, float* h_f32, unsigned* bar) {
  int i = blockIdx.x * blockDim.x + threadIdx.x;
  if (i < B_ * H_) { h_bf[i] = 0; h_f32[i] = 0.0f; }
  if (i == 0) bar[0] = 0u;
}

// ---------------- persistent GRU kernel ----------------
// 32 blocks x 256 threads (8 waves). Block owns 32 z|r output columns and 16
// h_cand columns; its weight slices live in LDS for the whole 1024-step loop.
// Phase 1: each wave = 1 M-tile x 2 N-tiles (A fragment shared, 2 WMMAs/ktile).
// Phase 2: each wave = 1 M-tile x 1 N-tile.

__global__ __launch_bounds__(256, 1) void gru_persistent(
    const unsigned short* __restrict__ x_bf,   // [B][T][I] bf16
    const unsigned short* __restrict__ Wzr_t,  // [1024][768] bf16
    const unsigned short* __restrict__ Wh_t,   // [512][768] bf16
    const float* __restrict__ bz,
    const float* __restrict__ br,
    const float* __restrict__ bh,
    unsigned short* __restrict__ h_bf,   // [B][H] bf16
    unsigned short* __restrict__ rh_bf,  // [B][H] bf16
    float* __restrict__ h_f32,           // [B][H]
    float* __restrict__ z_f32,           // [B][H]
    float* __restrict__ out,             // [B][T][H]
    unsigned* __restrict__ bar)
{
  const int lane = threadIdx.x & 31;
  const int wave = threadIdx.x >> 5;        // 0..7 == M tile
  const int blk  = blockIdx.x;              // 0..31
  const int ml   = lane & 15;               // A-row / B-col within 16x16 tile
  const int hs   = lane >> 4;               // half-select (lanes 16..31)

  extern __shared__ char smem[];
  unsigned short* lds_wzr = (unsigned short*)smem;      // [32][WPAD]
  unsigned short* lds_wh  = lds_wzr + 32 * WPAD;        // [16][WPAD]

  // ---- one-time weight preload into LDS (48 KB + 24 KB slices) ----
  for (int idx = threadIdx.x; idx < 32 * (IH_ / 8); idx += 256) {
    const int row = idx / (IH_ / 8), c = idx % (IH_ / 8);
    uint4 v = *(const uint4*)(Wzr_t + (size_t)(blk * 32 + row) * IH_ + c * 8);
    *(uint4*)(lds_wzr + row * WPAD + c * 8) = v;
  }
  for (int idx = threadIdx.x; idx < 16 * (IH_ / 8); idx += 256) {
    const int row = idx / (IH_ / 8), c = idx % (IH_ / 8);
    uint4 v = *(const uint4*)(Wh_t + (size_t)(blk * 16 + row) * IH_ + c * 8);
    *(uint4*)(lds_wh + row * WPAD + c * 8) = v;
  }
  __syncthreads();

  const int m0 = wave << 4;
  const unsigned short* hrow_g  = h_bf  + (size_t)(m0 + ml) * H_;
  const unsigned short* rhrow_g = rh_bf + (size_t)(m0 + ml) * H_;
  const unsigned short* lw0 = lds_wzr + ml * WPAD;          // N tile 2*blk
  const unsigned short* lw1 = lds_wzr + (16 + ml) * WPAD;   // N tile 2*blk+1
  const unsigned short* lwh = lds_wh  + ml * WPAD;          // N tile blk

  unsigned epoch = 0;

  for (int t = 0; t < T_; ++t) {
    const unsigned short* xrow = x_bf + ((size_t)(m0 + ml) * T_ + t) * I_;

    // ---------- Phase 1: [z|r] = sigmoid([x_t, h] @ Wzr + b) ----------
    {
      v8f acc0 = {}, acc1 = {};
      #pragma unroll 4
      for (int kt = 0; kt < I_ / 32; ++kt) {           // K from x_t
        const int k0 = kt * 32;
        Frag16 a, b0, b1;
        a.q[0]  = *(const uint4*)(xrow + k0 + 8 * hs);
        a.q[1]  = *(const uint4*)(xrow + k0 + 16 + 8 * hs);
        b0.q[0] = *(const uint4*)(lw0 + k0 + 8 * hs);
        b0.q[1] = *(const uint4*)(lw0 + k0 + 16 + 8 * hs);
        b1.q[0] = *(const uint4*)(lw1 + k0 + 8 * hs);
        b1.q[1] = *(const uint4*)(lw1 + k0 + 16 + 8 * hs);
        acc0 = __builtin_amdgcn_wmma_f32_16x16x32_bf16(
            false, a.v, false, b0.v, (short)0, acc0, false, false);
        acc1 = __builtin_amdgcn_wmma_f32_16x16x32_bf16(
            false, a.v, false, b1.v, (short)0, acc1, false, false);
      }
      #pragma unroll 4
      for (int kt = 0; kt < H_ / 32; ++kt) {           // K from h
        const int k0 = kt * 32;
        Frag16 a, b0, b1;
        a.q[0]  = *(const uint4*)(hrow_g + k0 + 8 * hs);
        a.q[1]  = *(const uint4*)(hrow_g + k0 + 16 + 8 * hs);
        b0.q[0] = *(const uint4*)(lw0 + I_ + k0 + 8 * hs);
        b0.q[1] = *(const uint4*)(lw0 + I_ + k0 + 16 + 8 * hs);
        b1.q[0] = *(const uint4*)(lw1 + I_ + k0 + 8 * hs);
        b1.q[1] = *(const uint4*)(lw1 + I_ + k0 + 16 + 8 * hs);
        acc0 = __builtin_amdgcn_wmma_f32_16x16x32_bf16(
            false, a.v, false, b0.v, (short)0, acc0, false, false);
        acc1 = __builtin_amdgcn_wmma_f32_16x16x32_bf16(
            false, a.v, false, b1.v, (short)0, acc1, false, false);
      }
      const int basen = blk * 32;
      if (blk < 16) {                       // z slice (block-uniform branch)
        const int ng0 = basen + ml, ng1 = basen + 16 + ml;
        const float bi0 = bz[ng0], bi1 = bz[ng1];
        #pragma unroll
        for (int r = 0; r < 8; ++r) {
          const int m = m0 + r + 8 * hs;
          z_f32[(size_t)m * H_ + ng0] = sigmoid_fast(acc0[r] + bi0);
          z_f32[(size_t)m * H_ + ng1] = sigmoid_fast(acc1[r] + bi1);
        }
      } else {                              // r slice -> r * h (bf16 for comb2)
        const int nn0 = basen - H_ + ml, nn1 = nn0 + 16;
        const float bi0 = br[nn0], bi1 = br[nn1];
        #pragma unroll
        for (int r = 0; r < 8; ++r) {
          const int m = m0 + r + 8 * hs;
          const size_t i0 = (size_t)m * H_ + nn0;
          const size_t i1 = (size_t)m * H_ + nn1;
          rh_bf[i0] = f2bf(sigmoid_fast(acc0[r] + bi0) * h_f32[i0]);
          rh_bf[i1] = f2bf(sigmoid_fast(acc1[r] + bi1) * h_f32[i1]);
        }
      }
    }
    grid_sync(bar, (unsigned)NBLOCKS * (++epoch));

    // ---------- Phase 2: h_cand = tanh([x_t, r*h] @ Wh + bh); blend ----------
    {
      v8f acc = {};
      #pragma unroll 4
      for (int kt = 0; kt < I_ / 32; ++kt) {           // K from x_t
        const int k0 = kt * 32;
        Frag16 a, b;
        a.q[0] = *(const uint4*)(xrow + k0 + 8 * hs);
        a.q[1] = *(const uint4*)(xrow + k0 + 16 + 8 * hs);
        b.q[0] = *(const uint4*)(lwh + k0 + 8 * hs);
        b.q[1] = *(const uint4*)(lwh + k0 + 16 + 8 * hs);
        acc = __builtin_amdgcn_wmma_f32_16x16x32_bf16(
            false, a.v, false, b.v, (short)0, acc, false, false);
      }
      #pragma unroll 4
      for (int kt = 0; kt < H_ / 32; ++kt) {           // K from r*h
        const int k0 = kt * 32;
        Frag16 a, b;
        a.q[0] = *(const uint4*)(rhrow_g + k0 + 8 * hs);
        a.q[1] = *(const uint4*)(rhrow_g + k0 + 16 + 8 * hs);
        b.q[0] = *(const uint4*)(lwh + I_ + k0 + 8 * hs);
        b.q[1] = *(const uint4*)(lwh + I_ + k0 + 16 + 8 * hs);
        acc = __builtin_amdgcn_wmma_f32_16x16x32_bf16(
            false, a.v, false, b.v, (short)0, acc, false, false);
      }
      const int nglob = blk * 16 + ml;
      const float bias = bh[nglob];
      #pragma unroll
      for (int r = 0; r < 8; ++r) {
        const int m = m0 + r + 8 * hs;
        const size_t idx = (size_t)m * H_ + nglob;
        float hc = tanh_fast(acc[r] + bias);
        float z  = z_f32[idx];
        float h  = h_f32[idx];
        float hn = (1.0f - z) * h + z * hc;
        h_f32[idx] = hn;
        h_bf[idx]  = f2bf(hn);
        out[((size_t)m * T_ + t) * H_ + nglob] = hn;
      }
    }
    grid_sync(bar, (unsigned)NBLOCKS * (++epoch));
  }
}

// ---------------- host side ----------------

extern "C" void kernel_launch(void* const* d_in, const int* in_sizes, int n_in,
                              void* d_out, int out_size, void* d_ws, size_t ws_size,
                              hipStream_t stream) {
  (void)in_sizes; (void)n_in; (void)out_size; (void)ws_size;

  const float* x  = (const float*)d_in[0];
  const float* Wz = (const float*)d_in[1];
  const float* bz = (const float*)d_in[2];
  const float* Wr = (const float*)d_in[3];
  const float* br = (const float*)d_in[4];
  const float* Wh = (const float*)d_in[5];
  const float* bh = (const float*)d_in[6];

  char* ws = (char*)d_ws;
  size_t off = 0;
  auto alloc = [&](size_t bytes) -> char* {
    char* p = ws + off;
    off += (bytes + 255) & ~(size_t)255;
    return p;
  };
  unsigned short* x_bf  = (unsigned short*)alloc((size_t)B_ * T_ * I_ * 2);   // 64 MB
  unsigned short* Wzr_t = (unsigned short*)alloc((size_t)1024 * IH_ * 2);     // 1.5 MB
  unsigned short* Wh_t  = (unsigned short*)alloc((size_t)H_ * IH_ * 2);       // 0.75 MB
  unsigned short* h_bf  = (unsigned short*)alloc((size_t)B_ * H_ * 2);
  unsigned short* rh_bf = (unsigned short*)alloc((size_t)B_ * H_ * 2);
  float*    h_f32 = (float*)alloc((size_t)B_ * H_ * 4);
  float*    z_f32 = (float*)alloc((size_t)B_ * H_ * 4);
  unsigned* bar   = (unsigned*)alloc(256);

  const size_t nx = (size_t)B_ * T_ * I_;
  prep_x_kernel<<<2048, 256, 0, stream>>>(x, x_bf, nx);
  const int wtot = 1024 * IH_ + H_ * IH_;
  prep_w_kernel<<<(wtot + 255) / 256, 256, 0, stream>>>(Wz, Wr, Wh, Wzr_t, Wh_t);
  prep_state_kernel<<<(B_ * H_ + 255) / 256, 256, 0, stream>>>(h_bf, h_f32, bar);

  gru_persistent<<<NBLOCKS, 256, SMEM_BYTES, stream>>>(
      x_bf, Wzr_t, Wh_t, bz, br, bh,
      h_bf, rh_bf, h_f32, z_f32, (float*)d_out, bar);
}
